// GrowableStateSpace_15745350107618
// MI455X (gfx1250) — matrix operand order repaired and checked by reading, not compiled, not run on hardware
//
#include <hip/hip_runtime.h>
#include <math.h>

// ---------------------------------------------------------------------------
// GrowableStateSpace on MI455X (gfx1250, wave32, WMMA 16x16x32 f16,
// async-to-LDS double-buffered staging).
//
//   q        = query @ Wq.T + bq                       [256,32]
//   scores   = (q @ pool.T) / sqrt(32)                 [256, 262144]
//   read_out = softmax(scores) @ pool                  [256,32]
//   argmin_n (p2[n] - 2*dot[n,b])  -> best_idx, min_dist
//   gate/sigmoid/append epilogue
//
// Compute-bound (~256 FLOP/B vs ~17 FLOP/B machine balance) -> all dot
// products via v_wmma_f32_16x16x32_f16 (K=32 exact fit). Per 32-row block
// per wave: 2 score WMMAs, 2 accumulation WMMAs, 1 ones-WMMA (softmax
// denominator). VALU epilogue minimized: exp2 arg comes straight out of the
// WMMA (scale folded into q), argmin is a branch-free v_and_or+v_min_u32
// chain on bias-positive key bits, lane-half swaps use immediate-pattern
// ds_swizzle (no address setup).
// ---------------------------------------------------------------------------

typedef __attribute__((ext_vector_type(16))) _Float16 v16h;
typedef __attribute__((ext_vector_type(8)))  _Float16 v8h;
typedef __attribute__((ext_vector_type(2)))  _Float16 v2h;
typedef __attribute__((ext_vector_type(8)))  float    v8f;
typedef __attribute__((ext_vector_type(8)))  unsigned v8u;

#define BQ        256
#define VD        32
#define QD        64
#define NPOOL     262144
#define BLK_ROWS  32
#define SCALE_LOG2E 0.2550348762731849f     // (1/sqrt(32)) * log2(e)
#define KC        (2.0f / SCALE_LOG2E)      // key = p2b - KC * dot'
#define P2BIAS    1024.0f                   // keeps key strictly positive

#if __has_builtin(__builtin_amdgcn_exp2f)
#define EXP2F __builtin_amdgcn_exp2f
#else
#define EXP2F exp2f
#endif

#define SHUF16(a, b) __builtin_shufflevector(a, b, 0,1,2,3,4,5,6,7,8,9,10,11,12,13,14,15)

// ds_swizzle group-of-32 patterns: offset = xor_mask<<10 | and_mask(0x1f)
#define SWZ_XOR16(x) __builtin_amdgcn_ds_swizzle((x), 0x401F)

template <int PAT>
__device__ __forceinline__ float swz_f(float v) {
    return __builtin_bit_cast(float,
        __builtin_amdgcn_ds_swizzle(__builtin_bit_cast(int, v), PAT));
}

// one per-lane 8-byte async global->LDS copy (ASYNCcnt-tracked)
__device__ __forceinline__ void async_copy_b64(const void* gptr, void* lptr) {
    unsigned lds = (unsigned)(uintptr_t)lptr;           // low 32 bits = LDS offset
    unsigned long long ga = (unsigned long long)(uintptr_t)gptr;
    asm volatile("global_load_async_to_lds_b64 %0, %1, off"
                 :: "v"(lds), "v"(ga) : "memory");
}
__device__ __forceinline__ void wait_async0() {
    asm volatile("s_wait_asynccnt 0" ::: "memory");
}

// ---------------------------------------------------------------------------
// k0: q = query @ Wq.T + bq ; new_vec = query @ Wn.T + bn
// ---------------------------------------------------------------------------
__global__ __launch_bounds__(256) void gss_k0(
    const float* __restrict__ query, const float* __restrict__ Wq,
    const float* __restrict__ bq,    const float* __restrict__ Wn,
    const float* __restrict__ bn,    float* __restrict__ qbuf,
    float* __restrict__ nbuf)
{
    int idx = blockIdx.x * 256 + threadIdx.x;      // [0, 8192)
    int b = idx >> 5, d = idx & 31;
    const float* qr = query + b * QD;
    const float* wq = Wq + d * QD;
    const float* wn = Wn + d * QD;
    float s1 = bq[d], s2 = bn[d];
#pragma unroll
    for (int k = 0; k < QD; ++k) {
        float x = qr[k];
        s1 = fmaf(x, wq[k], s1);
        s2 = fmaf(x, wn[k], s2);
    }
    qbuf[idx] = s1;
    nbuf[idx] = s2;
}

// ---------------------------------------------------------------------------
// k1: streaming WMMA pass over pool
// ---------------------------------------------------------------------------
__global__ __launch_bounds__(512) void gss_k1(
    const float* __restrict__ pool, const float* __restrict__ qbuf,
    float* __restrict__ lbuf, float* __restrict__ keybuf,
    int* __restrict__ idxbuf, float* __restrict__ accbuf,
    int blocks_per_wg)
{
    __shared__ float    Xbuf[2][BLK_ROWS * VD];   // async landing, f32
    __shared__ _Float16 lp16[BLK_ROWS * VD];      // pool block, f16 [n][d]
    __shared__ _Float16 lpT [BLK_ROWS * VD];      // transposed,  f16 [d][n]
    __shared__ float    p2s [BLK_ROWS];           // p2 + P2BIAS

    const int tid    = threadIdx.x;
    const int lane   = tid & 31;
    const int wv     = tid >> 5;          // wave id 0..15
    const int lane16 = lane & 15;
    const int h      = lane >> 4;         // lane-half
    const bool lo    = (h == 0);
    const int g      = blockIdx.x;
    const long base_row = (long)g * blocks_per_wg * BLK_ROWS;

    // ---- B matrix (qT * scale*log2e), once ---------------------------------
    const int qb = wv * 16;
    v16h Bq;
    {
        const float* qrow = qbuf + (qb + lane16) * VD + h * 16;
#pragma unroll
        for (int i = 0; i < 16; ++i) Bq[i] = (_Float16)(qrow[i] * SCALE_LOG2E);
    }
    // ---- all-ones A matrix for the softmax-denominator WMMA ---------------
    v16h Aones;
#pragma unroll
    for (int i = 0; i < 16; ++i) Aones[i] = (_Float16)1.0f;

    v8f acc0 = {};            // D[M=dim 0..15,  N=query]
    v8f acc1 = {};            // D[M=dim 16..31, N=query]
    v8f accL = {};            // D[M=*, N=query] = sum of exp (all rows equal)
    unsigned gmin = 0xFFFFFFFFu;   // packed (key bits | slot)
    int      gblk = 0;

    // prologue: async stage block 0
    async_copy_b64(pool + base_row * VD + tid * 2, &Xbuf[0][tid * 2]);

    for (int blk = 0; blk < blocks_per_wg; ++blk) {
        wait_async0();                       // this block's async copy done
        __syncthreads();                     // X visible; f16 bufs reusable

        // ---- cooperative convert: f32 -> f16 (row + transposed) + p2 ------
        {
            const float2 v = ((const float2*)Xbuf[blk & 1])[tid];
            v2h p; p[0] = (_Float16)v.x; p[1] = (_Float16)v.y;
            ((v2h*)lp16)[tid] = p;           // row-major f16
            const int n = tid >> 4, d0 = (tid & 15) * 2;
            lpT[(d0    ) * BLK_ROWS + n] = p[0];
            lpT[(d0 + 1) * BLK_ROWS + n] = p[1];
            float part = fmaf(v.x, v.x, v.y * v.y);
            part += swz_f<0x041F>(part);       // xor 1
            part += swz_f<0x081F>(part);       // xor 2
            part += swz_f<0x101F>(part);       // xor 4
            part += swz_f<0x201F>(part);       // xor 8
            if ((lane & 15) == 0) p2s[n] = part + P2BIAS;
        }
        __syncthreads();                     // f16 bufs + p2s ready

        // kick off next block's async copy (overlaps with compute below)
        if (blk + 1 < blocks_per_wg)
            async_copy_b64(pool + (base_row + (long)(blk + 1) * BLK_ROWS) * VD + tid * 2,
                           &Xbuf[(blk + 1) & 1][tid * 2]);

        // ---- score tiles: D[M=n, N=b] = pool_tile x qT' (pre-scaled) ------
        float t0w[8], t1w[8];
        unsigned bmin = 0xFFFFFFFFu;
#pragma unroll
        for (int t = 0; t < 2; ++t) {
            const _Float16* ap = lp16 + (t * 16 + lane16) * VD + h * 16;
            v8h alo = *(const v8h*)(ap);
            v8h ahi = *(const v8h*)(ap + 8);
            v16h A = SHUF16(alo, ahi);

            v8f zero = {};
            v8f D = __builtin_amdgcn_wmma_f32_16x16x32_f16(
                        false, A, false, Bq, (short)0, zero, false, false);

            const float4* p2v = (const float4*)p2s;
            float4 pa = p2v[t * 4 + 2 * h];
            float4 pb = p2v[t * 4 + 2 * h + 1];
            float pax[8] = {pa.x, pa.y, pa.z, pa.w, pb.x, pb.y, pb.z, pb.w};

            // D VGPR r: n = t*16 + r + 8*(L/16), b = qb + L%16
#pragma unroll
            for (int r = 0; r < 8; ++r) {
                float dotp = D[r];                       // dot * scale*log2e
                unsigned nl = (unsigned)(t * 16 + r + 8 * h);
                float keyp = fmaf(-KC, dotp, pax[r]);    // > 0 by construction
                unsigned kb = __builtin_bit_cast(unsigned, keyp);
                unsigned cand = (kb & 0xFFFFFFE0u) | nl; // v_and_or_b32
                bmin = (cand < bmin) ? cand : bmin;      // v_min_u32
                float w = EXP2F(dotp);
                if (t == 0) t0w[r] = w; else t1w[r] = w;
            }
        }
        if (bmin < gmin) { gmin = bmin; gblk = blk; }

        // ---- pack exp(score) into B layout (f16 first, then half-swap) ----
        v8u B2u;
#pragma unroll
        for (int j = 0; j < 4; ++j) {
            v2h a; a[0] = (_Float16)t0w[2 * j]; a[1] = (_Float16)t0w[2 * j + 1];
            v2h b; b[0] = (_Float16)t1w[2 * j]; b[1] = (_Float16)t1w[2 * j + 1];
            unsigned ua = __builtin_bit_cast(unsigned, a);
            unsigned ub = __builtin_bit_cast(unsigned, b);
            unsigned y  = lo ? ub : ua;
            unsigned sh = (unsigned)SWZ_XOR16((int)y);   // ds_swizzle imm
            B2u[j]     = lo ? ua : sh;      // elems 2j,2j+1   (K offs 0..7)
            B2u[4 + j] = lo ? sh : ub;      // elems 8+2j,...  (K offs 8..15)
        }
        v16h B2 = __builtin_bit_cast(v16h, B2u);

        // ---- acc[d,b] += poolT x B2 ; accL += ones x B2 -------------------
#pragma unroll
        for (int half = 0; half < 2; ++half) {
            const _Float16* tp = lpT + (half * 16 + lane16) * BLK_ROWS + h * 16;
            v8h alo = *(const v8h*)(tp);
            v8h ahi = *(const v8h*)(tp + 8);
            v16h A2 = SHUF16(alo, ahi);
            if (half == 0)
                acc0 = __builtin_amdgcn_wmma_f32_16x16x32_f16(
                           false, A2, false, B2, (short)0, acc0, false, false);
            else
                acc1 = __builtin_amdgcn_wmma_f32_16x16x32_f16(
                           false, A2, false, B2, (short)0, acc1, false, false);
        }
        accL = __builtin_amdgcn_wmma_f32_16x16x32_f16(
                   false, Aones, false, B2, (short)0, accL, false, false);
    }

    // ---- cross-lane-half reduction of the packed argmin -------------------
    {
        unsigned og = (unsigned)SWZ_XOR16((int)gmin);
        int      ob = SWZ_XOR16(gblk);
        if (og < gmin || (og == gmin && ob < gblk)) { gmin = og; gblk = ob; }
    }

    const int b = qb + lane16;
    if (lo) {
        lbuf[g * BQ + b]   = accL[0];   // sum of exp over this WG's chunk
        keybuf[g * BQ + b] =
            __builtin_bit_cast(float, gmin & 0xFFFFFFE0u) - P2BIAS;
        idxbuf[g * BQ + b] =
            (int)base_row + gblk * BLK_ROWS + (int)(gmin & 31u);
    }
    float* abase = accbuf + ((long)g * BQ + b) * VD;
#pragma unroll
    for (int r = 0; r < 8; ++r) {
        abase[r + 8 * h]      = acc0[r];
        abase[16 + r + 8 * h] = acc1[r];
    }
}

// ---------------------------------------------------------------------------
// k2: reduce per-WG partials + gate/sigmoid/append epilogue + outputs
// ---------------------------------------------------------------------------
__global__ __launch_bounds__(256) void gss_k2(
    const float* __restrict__ pool, const float* __restrict__ qbuf,
    const float* __restrict__ nbuf, const float* __restrict__ Wg,
    const float* __restrict__ bg,   const float* __restrict__ lbuf,
    const float* __restrict__ keybuf, const int* __restrict__ idxbuf,
    const float* __restrict__ accbuf, float* __restrict__ out, int nwg)
{
    const int b = threadIdx.x;
    float lsum = 0.f, mk = 3.0e38f;
    int   mi = 0;
    float acc[VD];
#pragma unroll
    for (int d = 0; d < VD; ++d) acc[d] = 0.f;

    for (int g = 0; g < nwg; ++g) {
        lsum += lbuf[g * BQ + b];
        float k = keybuf[g * BQ + b];
        int   i = idxbuf[g * BQ + b];
        if (k < mk || (k == mk && i < mi)) { mk = k; mi = i; }
        const float4* a = (const float4*)(accbuf + ((long)g * BQ + b) * VD);
#pragma unroll
        for (int j = 0; j < 8; ++j) {
            float4 v = a[j];
            acc[4 * j + 0] += v.x; acc[4 * j + 1] += v.y;
            acc[4 * j + 2] += v.z; acc[4 * j + 3] += v.w;
        }
    }

    float q[VD], q2 = 0.f;
#pragma unroll
    for (int d = 0; d < VD; ++d) { q[d] = qbuf[b * VD + d]; q2 = fmaf(q[d], q[d], q2); }

    const float invl = 1.0f / lsum;
    const float mind = sqrtf(fmaxf(mk + q2, 0.f));
    const bool append = (mind >= 1.5f);

    const float* bp = pool + (long)mi * VD;
    float gd = bg[0];
#pragma unroll
    for (int d = 0; d < VD; ++d) gd = fmaf(Wg[d], bp[d] - q[d], gd);
    const float gate = 1.0f / (1.0f + __expf(-gd));

#pragma unroll
    for (int d = 0; d < VD; ++d) {
        out[b * VD + d] = acc[d] * invl;                       // read_out
        float nv  = nbuf[b * VD + d];
        float upd = (1.0f - gate) * bp[d] + gate * nv;
        out[BQ * VD + b * VD + d] = append ? nv : upd;         // row
    }
    ((int*)out)[2 * BQ * VD + b] = mi;                         // best_idx
    out[2 * BQ * VD + BQ + b]    = append ? 1.0f : 0.0f;       // append
}

// ---------------------------------------------------------------------------
extern "C" void kernel_launch(void* const* d_in, const int* in_sizes, int n_in,
                              void* d_out, int out_size, void* d_ws, size_t ws_size,
                              hipStream_t stream)
{
    (void)in_sizes; (void)n_in; (void)out_size;
    const float* query = (const float*)d_in[0];
    const float* pool  = (const float*)d_in[1];
    const float* Wq    = (const float*)d_in[2];
    const float* bq    = (const float*)d_in[3];
    const float* Wg    = (const float*)d_in[4];
    const float* bg    = (const float*)d_in[5];
    const float* Wn    = (const float*)d_in[6];
    const float* bn    = (const float*)d_in[7];
    float* out = (float*)d_out;

    const size_t base_bytes   = 2u * BQ * VD * sizeof(float);        // q + new_vec
    const size_t per_wg_bytes = (size_t)BQ * (3 + VD) * sizeof(float);
    int nwg = 512;                                                   // divides 8192 blocks
    while (nwg > 1 && base_bytes + (size_t)nwg * per_wg_bytes > ws_size) nwg >>= 1;
    const int blocks_per_wg = (NPOOL / BLK_ROWS) / nwg;

    char* ws = (char*)d_ws;
    float* qbuf   = (float*)(ws);
    float* nbuf   = (float*)(ws + BQ * VD * sizeof(float));
    float* lbuf   = (float*)(ws + base_bytes);
    float* keybuf = lbuf + (size_t)nwg * BQ;
    int*   idxbuf = (int*)(keybuf + (size_t)nwg * BQ);
    float* accbuf = (float*)(idxbuf + (size_t)nwg * BQ);

    gss_k0<<<32, 256, 0, stream>>>(query, Wq, bq, Wn, bn, qbuf, nbuf);
    gss_k1<<<nwg, 512, 0, stream>>>(pool, qbuf, lbuf, keybuf, idxbuf, accbuf,
                                    blocks_per_wg);
    gss_k2<<<1, 256, 0, stream>>>(pool, qbuf, nbuf, Wg, bg, lbuf, keybuf,
                                  idxbuf, accbuf, out, nwg);
}
